// SelfAttentionBlock_13176959664393
// MI455X (gfx1250) — matrix operand order repaired and checked by reading, not compiled, not run on hardware
//
#include <hip/hip_runtime.h>
#include <hip/hip_bf16.h>
#include <math.h>

// ---------------------------------------------------------------------------
// SelfAttentionBlock for MI455X (gfx1250): bf16 WMMA GEMMs + fp32 epilogues.
// B=32, S=512, D=1024.  pooled = (colsum_q softmax(qk^T/sqrt(D))) @ v
// GEMM tiling: block 128x256 (8 waves), wave 64x64 = 4x4 WMMA fragments
// -> 16 v_wmma per K-step vs 16 vector loads.
// ---------------------------------------------------------------------------

#define BATCH 32
#define SEQ   512
#define DIM   1024
#define BS    (BATCH * SEQ)      // 16384 rows
#define SCALE 0.03125f           // 1/sqrt(1024)

typedef __bf16 bf16_t;
typedef __attribute__((ext_vector_type(16))) __bf16 v16bf;
typedef __attribute__((ext_vector_type(8)))  __bf16 v8bf;
typedef __attribute__((ext_vector_type(4)))  __bf16 v4bf;
typedef __attribute__((ext_vector_type(8)))  float  v8f;

// ---------------------------------------------------------------------------
// Fragment loaders matching the CDNA5 16-bit WMMA VGPR layouts (ISA 7.12.2).
// A (16x32, MxK): lanes 0-15 hold row M=lane, K[0..7] in v0-3 and K[16..23]
// in v4-7; lanes 16-31 hold K[8..15] and K[24..31].  => two 16-byte loads.
// B (32x16, KxN): lane holds column N=(lane&15), same K order as A:
// K[h*8..h*8+7] then K[16+h*8..16+h*8+7].  => two 16-byte loads.  Because we compute
// x @ W^T with row-major W[N,K], W's rows are exactly WMMA B-columns.
// ---------------------------------------------------------------------------
__device__ __forceinline__ v16bf load_a_frag(const bf16_t* A, int lda,
                                             int m0, int kk, int lane) {
    const int r = lane & 15;
    const int h = lane >> 4;
    const bf16_t* p = A + (size_t)(m0 + r) * lda + kk + h * 8;
    v8bf lo = *(const v8bf*)p;          // K[h*8 .. h*8+7]
    v8bf hi = *(const v8bf*)(p + 16);   // K[16+h*8 .. 16+h*8+7]
    v16bf a;
#pragma unroll
    for (int i = 0; i < 8; ++i) { a[i] = lo[i]; a[8 + i] = hi[i]; }
    return a;
}

__device__ __forceinline__ v16bf load_b_frag(const bf16_t* Bm, int ldb,
                                             int n0, int kk, int lane) {
    const int c = lane & 15;
    const int h = lane >> 4;
    const bf16_t* p = Bm + (size_t)(n0 + c) * ldb + kk + h * 8;
    v8bf lo = *(const v8bf*)p;          // K[h*8 .. h*8+7]
    v8bf hi = *(const v8bf*)(p + 16);   // K[16+h*8 .. 16+h*8+7]
    v16bf b;
#pragma unroll
    for (int i = 0; i < 8; ++i) { b[i] = lo[i]; b[8 + i] = hi[i]; }
    return b;
}

__device__ __forceinline__ v8f wmma_bf16(v16bf a, v16bf b, v8f c) {
    return __builtin_amdgcn_wmma_f32_16x16x32_bf16(
        /*neg_a=*/false, a, /*neg_b=*/false, b,
        /*c_mod=*/(short)0, c, /*reuse_a=*/false, /*reuse_b=*/false);
}

// 64x64 wave tile over K=DIM, A and B-as-rows both row-major with ld=DIM.
__device__ __forceinline__ void gemm_wave_64x64(const bf16_t* __restrict__ A,
                                                const bf16_t* __restrict__ Bm,
                                                int m0, int n0, int lane,
                                                v8f acc[4][4]) {
#pragma unroll
    for (int i = 0; i < 4; ++i)
#pragma unroll
        for (int j = 0; j < 4; ++j) acc[i][j] = (v8f){0,0,0,0,0,0,0,0};

    for (int kk = 0; kk < DIM; kk += 32) {
        v16bf a[4], b[4];
#pragma unroll
        for (int i = 0; i < 4; ++i) a[i] = load_a_frag(A, DIM, m0 + 16 * i, kk, lane);
#pragma unroll
        for (int j = 0; j < 4; ++j) b[j] = load_b_frag(Bm, DIM, n0 + 16 * j, kk, lane);
#pragma unroll
        for (int i = 0; i < 4; ++i)
#pragma unroll
            for (int j = 0; j < 4; ++j)
                acc[i][j] = wmma_bf16(a[i], b[j], acc[i][j]);
    }
}

// ---------------------------------------------------------------------------
// Kernel 1: fp32 -> bf16 conversion (4 elements per thread)
// ---------------------------------------------------------------------------
__global__ void k_cvt_bf16(const float* __restrict__ src,
                           bf16_t* __restrict__ dst, int n) {
    int i = (blockIdx.x * blockDim.x + threadIdx.x) * 4;
    if (i < n) {
        float4 f = *(const float4*)(src + i);
        v4bf o;
        o[0] = (bf16_t)f.x; o[1] = (bf16_t)f.y;
        o[2] = (bf16_t)f.z; o[3] = (bf16_t)f.w;
        *(v4bf*)(dst + i) = o;
    }
}

// ---------------------------------------------------------------------------
// Kernel 2: QKV projection GEMM (M=16384, N=1024, K=1024), z selects Q/K/V.
// Block: 256 threads = 8 waves (2 along M x 4 along N); block tile 128x256.
// Bias epilogue + bf16 store.
// ---------------------------------------------------------------------------
__global__ void __launch_bounds__(256)
k_qkv_gemm(const bf16_t* __restrict__ X,
           const bf16_t* __restrict__ Wq, const bf16_t* __restrict__ Wk,
           const bf16_t* __restrict__ Wv,
           const float* __restrict__ bq, const float* __restrict__ bk,
           const float* __restrict__ bv,
           bf16_t* __restrict__ Q, bf16_t* __restrict__ Kh,
           bf16_t* __restrict__ V) {
    const int z = blockIdx.z;
    const bf16_t* W   = (z == 0) ? Wq : (z == 1) ? Wk : Wv;
    const float*  bia = (z == 0) ? bq : (z == 1) ? bk : bv;
    bf16_t*       Out = (z == 0) ? Q  : (z == 1) ? Kh : V;

    const int tid   = threadIdx.x;
    const int lane  = tid & 31;
    const int wave  = tid >> 5;
    const int waveM = wave & 1;          // 2 waves along M
    const int waveN = wave >> 1;         // 4 waves along N
    const int m0 = blockIdx.x * 128 + waveM * 64;
    const int n0 = blockIdx.y * 256 + waveN * 64;

    v8f acc[4][4];
    gemm_wave_64x64(X, W, m0, n0, lane, acc);

    // C layout: VGPR i, lanes 0-15 -> row base+i, col lane&15;
    //           lanes 16-31 -> row base+8+i.
#pragma unroll
    for (int mi = 0; mi < 4; ++mi) {
#pragma unroll
        for (int ni = 0; ni < 4; ++ni) {
            const int col = n0 + ni * 16 + (lane & 15);
            const int row = m0 + mi * 16 + (lane >> 4) * 8;
            const float bc = bia[col];
#pragma unroll
            for (int i = 0; i < 8; ++i)
                Out[(size_t)(row + i) * DIM + col] = (bf16_t)(acc[mi][ni][i] + bc);
        }
    }
}

// ---------------------------------------------------------------------------
// Kernel 3: per-batch scores GEMM  S[q][k] = (q . k) / 32
// M=N=512, K=1024 per batch; grid (4, 2, 32); fp32 store to workspace.
// ---------------------------------------------------------------------------
__global__ void __launch_bounds__(256)
k_scores_gemm(const bf16_t* __restrict__ Q, const bf16_t* __restrict__ Kh,
              float* __restrict__ Sc) {
    const int b = blockIdx.z;
    const bf16_t* Qb = Q  + (size_t)b * SEQ * DIM;
    const bf16_t* Kb = Kh + (size_t)b * SEQ * DIM;
    float*        Sb = Sc + (size_t)b * SEQ * SEQ;

    const int tid   = threadIdx.x;
    const int lane  = tid & 31;
    const int wave  = tid >> 5;
    const int waveM = wave & 1;
    const int waveN = wave >> 1;
    const int m0 = blockIdx.x * 128 + waveM * 64;
    const int n0 = blockIdx.y * 256 + waveN * 64;

    v8f acc[4][4];
    gemm_wave_64x64(Qb, Kb, m0, n0, lane, acc);   // k rows = WMMA B columns

#pragma unroll
    for (int mi = 0; mi < 4; ++mi) {
#pragma unroll
        for (int ni = 0; ni < 4; ++ni) {
            const int col = n0 + ni * 16 + (lane & 15);
            const int row = m0 + mi * 16 + (lane >> 4) * 8;
#pragma unroll
            for (int i = 0; i < 8; ++i)
                Sb[(size_t)(row + i) * SEQ + col] = acc[mi][ni][i] * SCALE;
        }
    }
}

// ---------------------------------------------------------------------------
// Kernel 4: masked row softmax, in place.  One wave32 per row (512 keys ->
// 16 values per lane), wave shuffles for max / sum reductions.
// ---------------------------------------------------------------------------
__global__ void __launch_bounds__(128)
k_softmax_rows(float* __restrict__ Sc, const int* __restrict__ mask) {
    const int row  = blockIdx.x * (blockDim.x >> 5) + (threadIdx.x >> 5);
    const int lane = threadIdx.x & 31;
    const int b = row >> 9;       // /SEQ
    float* srow = Sc + (size_t)row * SEQ;
    const int* mrow = mask + (size_t)b * SEQ;

    float vals[16];
    float mx = -INFINITY;
#pragma unroll
    for (int j = 0; j < 16; ++j) {
        const int k = lane + j * 32;
        float s = mrow[k] ? srow[k] : -INFINITY;
        vals[j] = s;
        mx = fmaxf(mx, s);
    }
#pragma unroll
    for (int off = 16; off; off >>= 1) mx = fmaxf(mx, __shfl_xor(mx, off, 32));

    float sum = 0.f;
#pragma unroll
    for (int j = 0; j < 16; ++j) {
        float e = __expf(vals[j] - mx);
        vals[j] = e;
        sum += e;
    }
#pragma unroll
    for (int off = 16; off; off >>= 1) sum += __shfl_xor(sum, off, 32);

    const float inv = 1.0f / sum;
#pragma unroll
    for (int j = 0; j < 16; ++j) srow[lane + j * 32] = vals[j] * inv;
}

// ---------------------------------------------------------------------------
// Kernel 5: colsum[b][k] = sum_q weights[b][q][k]
// ---------------------------------------------------------------------------
__global__ void __launch_bounds__(256)
k_colsum(const float* __restrict__ Sc, float* __restrict__ cs) {
    const int b = blockIdx.x;
    const int k = blockIdx.y * 256 + threadIdx.x;
    const float* base = Sc + (size_t)b * SEQ * SEQ + k;
    float acc = 0.f;
#pragma unroll 4
    for (int q = 0; q < SEQ; ++q) acc += base[(size_t)q * SEQ];
    cs[b * SEQ + k] = acc;
}

// ---------------------------------------------------------------------------
// Kernel 6: pooled[b][d] = sum_k colsum[b][k] * v[b][k][d]
// ---------------------------------------------------------------------------
__global__ void __launch_bounds__(256)
k_pooled(const float* __restrict__ cs, const bf16_t* __restrict__ V,
         float* __restrict__ out) {
    const int b = blockIdx.x;
    const int d = blockIdx.y * 256 + threadIdx.x;
    const bf16_t* vb = V + (size_t)b * SEQ * DIM;
    const float*  c  = cs + b * SEQ;
    float acc = 0.f;
#pragma unroll 4
    for (int k = 0; k < SEQ; ++k)
        acc += c[k] * (float)vb[(size_t)k * DIM + d];
    out[(size_t)b * DIM + d] = acc;
}

// ---------------------------------------------------------------------------
// Host-side launcher
// ---------------------------------------------------------------------------
extern "C" void kernel_launch(void* const* d_in, const int* in_sizes, int n_in,
                              void* d_out, int out_size, void* d_ws, size_t ws_size,
                              hipStream_t stream) {
    const float* x    = (const float*)d_in[0];   // [B,S,D]
    const int*   mask = (const int*)  d_in[1];   // [B,S]
    const float* Wq   = (const float*)d_in[2];
    const float* bq   = (const float*)d_in[3];
    const float* Wk   = (const float*)d_in[4];
    const float* bk   = (const float*)d_in[5];
    const float* Wv   = (const float*)d_in[6];
    const float* bv   = (const float*)d_in[7];
    float* out = (float*)d_out;                  // [B,D]

    // workspace layout (bytes)
    char* w = (char*)d_ws;
    const size_t szX  = (size_t)BS * DIM * sizeof(bf16_t);         // 32 MB
    const size_t szW  = (size_t)DIM * DIM * sizeof(bf16_t);        //  2 MB each
    const size_t szSc = (size_t)BATCH * SEQ * SEQ * sizeof(float); // 32 MB
    bf16_t* xh  = (bf16_t*)(w);
    bf16_t* wqh = (bf16_t*)(w + szX);
    bf16_t* wkh = (bf16_t*)(w + szX + szW);
    bf16_t* wvh = (bf16_t*)(w + szX + 2 * szW);
    bf16_t* qh  = (bf16_t*)(w + szX + 3 * szW);
    bf16_t* kh  = (bf16_t*)(w + 2 * szX + 3 * szW);
    bf16_t* vh  = (bf16_t*)(w + 3 * szX + 3 * szW);
    float*  sc  = (float*) (w + 4 * szX + 3 * szW);
    float*  cs  = (float*) (w + 4 * szX + 3 * szW + szSc);

    // 1) convert inputs to bf16
    {
        int nX = BS * DIM;                       // 16.7M
        k_cvt_bf16<<<nX / (4 * 256), 256, 0, stream>>>(x, xh, nX);
        int nW = DIM * DIM;                      // 1M each
        k_cvt_bf16<<<nW / (4 * 256), 256, 0, stream>>>(Wq, wqh, nW);
        k_cvt_bf16<<<nW / (4 * 256), 256, 0, stream>>>(Wk, wkh, nW);
        k_cvt_bf16<<<nW / (4 * 256), 256, 0, stream>>>(Wv, wvh, nW);
    }

    // 2) QKV projection: [16384,1024] x [1024,1024]^T, z = {q,k,v}
    {
        dim3 grid(BS / 128, DIM / 256, 3);
        k_qkv_gemm<<<grid, 256, 0, stream>>>(xh, wqh, wkh, wvh,
                                             bq, bk, bv, qh, kh, vh);
    }

    // 3) scores = q k^T / 32, per batch
    {
        dim3 grid(SEQ / 128, SEQ / 256, BATCH);
        k_scores_gemm<<<grid, 256, 0, stream>>>(qh, kh, sc);
    }

    // 4) masked row softmax (in place): 16384 rows, 4 waves per block
    k_softmax_rows<<<(BATCH * SEQ) / 4, 128, 0, stream>>>(sc, mask);

    // 5) column sums of weights per batch
    {
        dim3 grid(BATCH, SEQ / 256);
        k_colsum<<<grid, 256, 0, stream>>>(sc, cs);
    }

    // 6) pooled = colsum @ v
    {
        dim3 grid(BATCH, DIM / 256);
        k_pooled<<<grid, 256, 0, stream>>>(cs, vh, out);
    }
}
